// CosLoss_7241314861436
// MI455X (gfx1250) — matrix operand order repaired and checked by reading, not compiled, not run on hardware
//
#include <hip/hip_runtime.h>
#include <hip/hip_bf16.h>

typedef _Float16 v16h __attribute__((ext_vector_type(16)));
typedef _Float16 v8h  __attribute__((ext_vector_type(8)));
typedef _Float16 v4h  __attribute__((ext_vector_type(4)));
typedef float    v8f  __attribute__((ext_vector_type(8)));

#define NQ      8192   // combined rows per tensor (8*32*32)
#define FDIM    1024   // 4*4*64
#define MSPLIT  4      // codebook splits across blockIdx.y
#define MCHUNK  (NQ / MSPLIT)   // 2048 cols per split

// feature f (multiple of 4) of combined row q -> flat element index in (8,128,128,64)
__device__ __forceinline__ int flat_index(int q, int f4) {
    int b  = q >> 10;
    int wb = (q >> 5) & 31;
    int hb = q & 31;
    int a   = f4 >> 8;       // w offset within block
    int rem = f4 & 255;      // bb*64 + d (contiguous in memory)
    return (((b * 128 + wb * 4 + a) * 128 + hb * 4) << 6) + rem;
}

// ---------------------------------------------------------------------------
// Kernel 1: combine + normalize. blockIdx.x in [0,16384): first 8192 = x rows,
// next 8192 = y rows. 256 threads, 4 features each.
// ---------------------------------------------------------------------------
__global__ void normalize_kernel(const float* __restrict__ x,
                                 const float* __restrict__ y,
                                 _Float16* __restrict__ xh,
                                 _Float16* __restrict__ yh,
                                 float* __restrict__ ymean,
                                 float* __restrict__ yinv) {
    const int t   = threadIdx.x;
    const int q   = blockIdx.x & (NQ - 1);
    const bool isY = blockIdx.x >= NQ;
    const float* src = isY ? y : x;
    _Float16*   dst = isY ? yh : xh;

    const int f4 = t * 4;
    float4 v = *(const float4*)(src + flat_index(q, f4));

    float sum = v.x + v.y + v.z + v.w;
    float sq  = v.x * v.x + v.y * v.y + v.z * v.z + v.w * v.w;
    #pragma unroll
    for (int off = 16; off; off >>= 1) {
        sum += __shfl_xor(sum, off, 32);
        sq  += __shfl_xor(sq,  off, 32);
    }
    __shared__ float s_sum[8], s_sq[8], s_mi[2];
    const int wid = t >> 5, ln = t & 31;
    if (ln == 0) { s_sum[wid] = sum; s_sq[wid] = sq; }
    __syncthreads();
    if (t == 0) {
        float S = 0.f, Q2 = 0.f;
        #pragma unroll
        for (int i = 0; i < 8; ++i) { S += s_sum[i]; Q2 += s_sq[i]; }
        float mean = S * (1.0f / FDIM);
        float var  = Q2 - S * S * (1.0f / FDIM);   // sum of centered squares
        float inv  = 1.0f / (sqrtf(fmaxf(var, 0.0f)) + 1e-5f);
        s_mi[0] = mean; s_mi[1] = inv;
        if (isY) { ymean[q] = mean; yinv[q] = inv; }
    }
    __syncthreads();
    const float mean = s_mi[0], inv = s_mi[1];

    v4h o;
    o[0] = (_Float16)((v.x - mean) * inv);
    o[1] = (_Float16)((v.y - mean) * inv);
    o[2] = (_Float16)((v.z - mean) * inv);
    o[3] = (_Float16)((v.w - mean) * inv);
    *(v4h*)(dst + (size_t)q * FDIM + f4) = o;
}

// ---------------------------------------------------------------------------
// Kernel 2: fused GEMM (xn @ yn^T) + per-row max/argmax via WMMA f16->f32.
// grid (32, MSPLIT) x 256 threads (8 waves). Wave w owns 32 query rows
// [blk.x*256 + w*32, +32) as two 16-row A tiles; each LDS panel holds 32
// codebook rows; the wave computes a 32x32 block with 4 accumulators so one
// A load feeds two WMMAs. K loop kept rolled (unroll 2) so A fragments are
// NOT hoisted/spilled across the panel loop.
// ---------------------------------------------------------------------------
__global__ void simmax_kernel(const _Float16* __restrict__ xh,
                              const _Float16* __restrict__ yh,
                              float* __restrict__ rmax_part,
                              int*   __restrict__ rarg_part) {
    __shared__ _Float16 Bs[32 * FDIM];     // 64 KB

    const int t    = threadIdx.x;
    const int lane = t & 31;
    const int wave = t >> 5;
    const int hi   = lane >> 4;            // 0: lanes 0-15, 1: lanes 16-31
    const int c    = lane & 15;            // B column / A row within half
    const int nbase = blockIdx.x * 256 + wave * 32;
    const int mbase0 = blockIdx.y * MCHUNK;

    // A fragment bases for the two 16-row tiles (ISA A layout: lane half
    // selects K sub-range; per lane two contiguous 16B chunks per k-step).
    const _Float16* aptr0 = xh + (size_t)(nbase + c) * FDIM + hi * 8;
    const _Float16* aptr1 = aptr0 + (size_t)16 * FDIM;

    float vmax[2][8];
    int   varg[2][8];
    #pragma unroll
    for (int g = 0; g < 2; ++g)
        #pragma unroll
        for (int r = 0; r < 8; ++r) { vmax[g][r] = -3.0e38f; varg[g][r] = 0; }

    for (int mo = 0; mo < MCHUNK / 32; ++mo) {
        __syncthreads();
        // stage 32 codebook rows (contiguous 64KB) into LDS
        {
            const uint4* s = (const uint4*)(yh + (size_t)(mbase0 + mo * 32) * FDIM);
            uint4*       d = (uint4*)Bs;
            #pragma unroll
            for (int i = t; i < 4096; i += 256) d[i] = s[i];
        }
        __syncthreads();

        v8f acc00 = {}, acc01 = {}, acc10 = {}, acc11 = {};
        const _Float16* bptr0 = Bs + (size_t)c * FDIM + hi * 8;          // cols +0..15
        const _Float16* bptr1 = bptr0 + (size_t)16 * FDIM;               // cols +16..31

        #pragma unroll 2
        for (int kt = 0; kt < FDIM / 32; ++kt) {
            const int kb = kt * 32;
            union { v16h v; v8h h[2]; } A0, A1;
            A0.h[0] = *(const v8h*)(aptr0 + kb);
            A0.h[1] = *(const v8h*)(aptr0 + kb + 16);
            A1.h[0] = *(const v8h*)(aptr1 + kb);
            A1.h[1] = *(const v8h*)(aptr1 + kb + 16);
            v16h B0 = __builtin_shufflevector(*(const v8h*)(bptr0 + kb), *(const v8h*)(bptr0 + kb + 16), 0, 1, 2, 3, 4, 5, 6, 7, 8, 9, 10, 11, 12, 13, 14, 15);
            v16h B1 = __builtin_shufflevector(*(const v8h*)(bptr1 + kb), *(const v8h*)(bptr1 + kb + 16), 0, 1, 2, 3, 4, 5, 6, 7, 8, 9, 10, 11, 12, 13, 14, 15);
            acc00 = __builtin_amdgcn_wmma_f32_16x16x32_f16(false, A0.v, false, B0, (short)0, acc00, false, false);
            acc01 = __builtin_amdgcn_wmma_f32_16x16x32_f16(false, A0.v, false, B1, (short)0, acc01, false, false);
            acc10 = __builtin_amdgcn_wmma_f32_16x16x32_f16(false, A1.v, false, B0, (short)0, acc10, false, false);
            acc11 = __builtin_amdgcn_wmma_f32_16x16x32_f16(false, A1.v, false, B1, (short)0, acc11, false, false);
        }

        const int col0 = mbase0 + mo * 32 + c;
        const int col1 = col0 + 16;
        #pragma unroll
        for (int r = 0; r < 8; ++r) {
            // check smaller column first -> first-max tie-break preserved
            if (acc00[r] > vmax[0][r]) { vmax[0][r] = acc00[r]; varg[0][r] = col0; }
            if (acc01[r] > vmax[0][r]) { vmax[0][r] = acc01[r]; varg[0][r] = col1; }
            if (acc10[r] > vmax[1][r]) { vmax[1][r] = acc10[r]; varg[1][r] = col0; }
            if (acc11[r] > vmax[1][r]) { vmax[1][r] = acc11[r]; varg[1][r] = col1; }
        }
    }

    // reduce max/argmax across the 16 lanes of each half (columns disjoint per
    // lane; prefer smaller index on exact ties = argmax-first semantics)
    #pragma unroll
    for (int g = 0; g < 2; ++g) {
        #pragma unroll
        for (int r = 0; r < 8; ++r) {
            float v  = vmax[g][r];
            int   id = varg[g][r];
            #pragma unroll
            for (int m = 1; m < 16; m <<= 1) {
                float ov = __shfl_xor(v,  m, 32);
                int   oi = __shfl_xor(id, m, 32);
                if (ov > v || (ov == v && oi < id)) { v = ov; id = oi; }
            }
            if (c == 0) {
                const int row = nbase + g * 16 + hi * 8 + r;
                rmax_part[blockIdx.y * NQ + row] = v;
                rarg_part[blockIdx.y * NQ + row] = id;
            }
        }
    }
}

// ---------------------------------------------------------------------------
// Kernel 3: merge the MSPLIT partial max/argmax (parts ordered by column
// range, so strict > in part order keeps the first maximum).
// ---------------------------------------------------------------------------
__global__ void merge_kernel(const float* __restrict__ rmax_part,
                             const int*   __restrict__ rarg_part,
                             float* __restrict__ rowmax,
                             int*   __restrict__ rowarg) {
    const int row = blockIdx.x * 256 + threadIdx.x;
    float v = -3.0e38f;
    int   id = 0;
    #pragma unroll
    for (int p = 0; p < MSPLIT; ++p) {
        float pv = rmax_part[p * NQ + row];
        if (pv > v) { v = pv; id = rarg_part[p * NQ + row]; }
    }
    rowmax[row] = v;
    rowarg[row] = id;
}

// ---------------------------------------------------------------------------
// Kernel 4: cosloss = mean(1 - rowmax)
// ---------------------------------------------------------------------------
__global__ void loss_kernel(const float* __restrict__ rowmax,
                            float* __restrict__ out) {
    const int t = threadIdx.x;
    float s = 0.f;
    for (int i = t; i < NQ; i += 256) s += 1.0f - rowmax[i];
    #pragma unroll
    for (int off = 16; off; off >>= 1) s += __shfl_xor(s, off, 32);
    __shared__ float sp[8];
    if ((t & 31) == 0) sp[t >> 5] = s;
    __syncthreads();
    if (t == 0) {
        float tot = 0.f;
        #pragma unroll
        for (int i = 0; i < 8; ++i) tot += sp[i];
        out[0] = tot * (1.0f / NQ);
    }
}

// ---------------------------------------------------------------------------
// Kernel 5: new_x = normalized y row [argmax], scattered back to (8,128,128,64).
// Recompute fp32 normalization from original y using stored mean/inv.
// new_x starts at d_out+1 (4B misaligned for float4) -> scalar stores.
// ---------------------------------------------------------------------------
__global__ void gather_kernel(const float* __restrict__ y,
                              const int*   __restrict__ rowarg,
                              const float* __restrict__ ymean,
                              const float* __restrict__ yinv,
                              float* __restrict__ out) {
    const int q = blockIdx.x;
    const int t = threadIdx.x;
    const int m = rowarg[q];
    const float mean = ymean[m];
    const float inv  = yinv[m];
    const int f4 = t * 4;
    float4 v = *(const float4*)(y + flat_index(m, f4));
    float* o = out + 1 + flat_index(q, f4);
    o[0] = (v.x - mean) * inv;
    o[1] = (v.y - mean) * inv;
    o[2] = (v.z - mean) * inv;
    o[3] = (v.w - mean) * inv;
}

extern "C" void kernel_launch(void* const* d_in, const int* in_sizes, int n_in,
                              void* d_out, int out_size, void* d_ws, size_t ws_size,
                              hipStream_t stream) {
    (void)in_sizes; (void)n_in; (void)out_size; (void)ws_size;
    const float* x = (const float*)d_in[0];
    const float* y = (const float*)d_in[1];
    float* out = (float*)d_out;

    // workspace layout
    char* ws = (char*)d_ws;
    _Float16* xh    = (_Float16*)(ws);                               // 16 MB
    _Float16* yh    = (_Float16*)(ws + (size_t)NQ * FDIM * 2);       // 16 MB
    float*    ymean = (float*)(ws + (size_t)NQ * FDIM * 4);          // 32 KB
    float*    yinvn = ymean + NQ;                                    // 32 KB
    float*    rmax  = yinvn + NQ;                                    // 32 KB
    int*      rarg  = (int*)(rmax + NQ);                             // 32 KB
    float*    rmaxp = (float*)(rarg + NQ);                           // 128 KB
    int*      rargp = (int*)(rmaxp + MSPLIT * NQ);                   // 128 KB

    normalize_kernel<<<2 * NQ, 256, 0, stream>>>(x, y, xh, yh, ymean, yinvn);
    simmax_kernel<<<dim3(NQ / 256, MSPLIT), 256, 0, stream>>>(xh, yh, rmaxp, rargp);
    merge_kernel<<<NQ / 256, 256, 0, stream>>>(rmaxp, rargp, rmax, rarg);
    loss_kernel<<<1, 256, 0, stream>>>(rmax, out);
    gather_kernel<<<NQ, 256, 0, stream>>>(y, rarg, ymean, yinvn, out);
}
